// AsymmInBatchNegativeInfoNCELoss_81088982548552
// MI455X (gfx1250) — compile-verified, matrix operand order checked by reading, not run on hardware
//
#include <hip/hip_runtime.h>
#include <hip/hip_bf16.h>

// ---------------------------------------------------------------------------
// InfoNCE (in-batch negatives) loss, fused:
//   1) normalize rows of q,p (f32 -> f16 in workspace)
//   2) S = qn*pn^T / temp computed tile-wise with v_wmma_f32_16x16x32_f16,
//      logsumexp streamed online (never materialize 8192x8192 scores)
//   3) mean(lse - diag) reduction
//
// GEMM structure (per 512-thread block = 16 waves, 32-row stripe in LDS):
//   - Each wave owns a 2x2 grid of 16x16 tiles (32x32 output): per K-chunk,
//     2 A ds-loads + 2 B global-loads feed FOUR independent WMMAs, so each
//     load-group drain is amortized over 4 matrix ops.
//   - 32-row stripe halves redundant B reads vs 16-row (L2 traffic ~3 GB).
//   - Moderate VGPR footprint (~170) keeps occupancy high so the scheduler's
//     wait-then-wmma pattern is hidden across waves.
// ---------------------------------------------------------------------------

typedef __attribute__((ext_vector_type(16))) _Float16 v16h;
typedef __attribute__((ext_vector_type(8)))  float    v8f;

#define DIMK    768
#define KCHUNKS 24          // DIMK / 32 (K per wmma_f32_16x16x32_f16)
#define GWAVES  16          // waves per workgroup in the GEMM kernel
#define MROWS   32          // rows per block (2 row-tiles)

union V16H {
    v16h  v;
    uint4 u[2];
};

union H4 {
    _Float16 h[4];
    uint2    u;
};

// ---------------------------------------------------------------------------
// Kernel 1: row L2-normalize + convert to fp16.
// One wave32 per row; rows [0,N) -> q, rows [N,2N) -> p.
// ---------------------------------------------------------------------------
__global__ __launch_bounds__(256) void normalize_f16_kernel(
    const float* __restrict__ q, const float* __restrict__ p,
    _Float16* __restrict__ qn, _Float16* __restrict__ pn, int N)
{
    const int wave = threadIdx.x >> 5;
    const int lane = threadIdx.x & 31;
    const int r    = blockIdx.x * 8 + wave;

    const float* __restrict__ src;
    _Float16*    __restrict__ dst;
    if (r < N) { src = q + (size_t)r * DIMK;        dst = qn + (size_t)r * DIMK; }
    else       { src = p + (size_t)(r - N) * DIMK;  dst = pn + (size_t)(r - N) * DIMK; }

    float4 vals[6];
    float  ss = 0.0f;
#pragma unroll
    for (int c = 0; c < 6; ++c) {
        vals[c] = *(const float4*)(src + c * 128 + lane * 4);
        ss += vals[c].x * vals[c].x + vals[c].y * vals[c].y
            + vals[c].z * vals[c].z + vals[c].w * vals[c].w;
    }
#pragma unroll
    for (int mask = 16; mask >= 1; mask >>= 1)
        ss += __shfl_xor(ss, mask, 32);

    const float scale = 1.0f / fmaxf(sqrtf(ss), 1e-8f);   // torch eps semantics

#pragma unroll
    for (int c = 0; c < 6; ++c) {
        H4 h4;
        h4.h[0] = (_Float16)(vals[c].x * scale);
        h4.h[1] = (_Float16)(vals[c].y * scale);
        h4.h[2] = (_Float16)(vals[c].z * scale);
        h4.h[3] = (_Float16)(vals[c].w * scale);
        *(uint2*)(dst + c * 128 + lane * 4) = h4.u;
    }
}

// ---------------------------------------------------------------------------
// Kernel 2: fused GEMM + online logsumexp + diag capture.
// ---------------------------------------------------------------------------
__global__ __launch_bounds__(512) void gemm_lse_kernel(
    const _Float16* __restrict__ qn, const _Float16* __restrict__ pn,
    const float* __restrict__ temp, float* __restrict__ rowloss, int N)
{
    __shared__ _Float16 As[MROWS * DIMK];     // 48 KB A stripe (one-shot)
    __shared__ float red_m[GWAVES][MROWS];
    __shared__ float red_s[GWAVES][MROWS];
    __shared__ float red_d[GWAVES][MROWS];

    const int tid    = threadIdx.x;
    const int wave   = tid >> 5;
    const int lane   = tid & 31;
    const int m_base = blockIdx.x * MROWS;
    const float invT = 1.0f / temp[0];

    // Cooperatively load A stripe (32 x 768 fp16 = 3072 uint4) into LDS.
    {
        const uint4* src = (const uint4*)(qn + (size_t)m_base * DIMK);
        uint4*       dst = (uint4*)As;
#pragma unroll
        for (int i = 0; i < (MROWS * DIMK) / (8 * 512); ++i)
            dst[tid + i * 512] = src[tid + i * 512];
    }
    __syncthreads();

    const int half = lane >> 4;        // C/D: row = v + 8*half (ISA layout)
    const int l16  = lane & 15;
    const int koff = half * 8;         // A/B 16-bit operand K sub-offset

    // A operand pointers for the two row-tiles (from LDS).
    const _Float16* a_lane0 = As + l16 * DIMK + koff;
    const _Float16* a_lane1 = As + (16 + l16) * DIMK + koff;

    // Per-lane online-softmax state: 2 row-tiles x 8 rows each.
    float m_run[2][8], s_run[2][8], diag[2][8];
#pragma unroll
    for (int rt = 0; rt < 2; ++rt)
#pragma unroll
        for (int v = 0; v < 8; ++v) {
            m_run[rt][v] = -3.0e38f; s_run[rt][v] = 0.0f; diag[rt][v] = 0.0f;
        }

    const int npairs = N >> 5;                       // pairs of 16-wide tiles

    for (int pp = wave; pp < npairs; pp += GWAVES) {
        const int n_base0 = pp << 5;                 // first tile of the pair
        const int n_base1 = n_base0 + 16;            // second tile of the pair
        const _Float16* b_lane0 = pn + (size_t)(n_base0 + l16) * DIMK + koff;
        const _Float16* b_lane1 = b_lane0 + (size_t)16 * DIMK;

        // Prefetch this lane's B rows for the next pair (global_prefetch_b8).
        if (pp + GWAVES < npairs) {
            __builtin_prefetch(b_lane0 + (size_t)GWAVES * 32 * DIMK, 0, 1);
            __builtin_prefetch(b_lane1 + (size_t)GWAVES * 32 * DIMK, 0, 1);
        }

        v8f acc00 = {}, acc01 = {}, acc10 = {}, acc11 = {};

#pragma unroll
        for (int kc = 0; kc < KCHUNKS; ++kc) {
            const int k0 = kc * 32;
            V16H a0, a1, b0, b1;
            b0.u[0] = *(const uint4*)(b_lane0 + k0);
            b0.u[1] = *(const uint4*)(b_lane0 + k0 + 16);
            b1.u[0] = *(const uint4*)(b_lane1 + k0);
            b1.u[1] = *(const uint4*)(b_lane1 + k0 + 16);
            a0.u[0] = *(const uint4*)(a_lane0 + k0);
            a0.u[1] = *(const uint4*)(a_lane0 + k0 + 16);
            a1.u[0] = *(const uint4*)(a_lane1 + k0);
            a1.u[1] = *(const uint4*)(a_lane1 + k0 + 16);

            // Four independent WMMAs per operand fetch group.
            acc00 = __builtin_amdgcn_wmma_f32_16x16x32_f16(
                false, a0.v, false, b0.v, (short)0, acc00, false, false);
            acc01 = __builtin_amdgcn_wmma_f32_16x16x32_f16(
                false, a0.v, false, b1.v, (short)0, acc01, false, false);
            acc10 = __builtin_amdgcn_wmma_f32_16x16x32_f16(
                false, a1.v, false, b0.v, (short)0, acc10, false, false);
            acc11 = __builtin_amdgcn_wmma_f32_16x16x32_f16(
                false, a1.v, false, b1.v, (short)0, acc11, false, false);
        }

        // Online update with scaled scores; 1 transcendental per element.
        const int col0 = n_base0 + l16;
        const int col1 = n_base1 + l16;
#pragma unroll
        for (int rt = 0; rt < 2; ++rt) {
#pragma unroll
            for (int v = 0; v < 8; ++v) {
                const int row = m_base + rt * 16 + v + 8 * half;
                const float x0 = (rt ? acc10[v] : acc00[v]) * invT;
                const float x1 = (rt ? acc11[v] : acc01[v]) * invT;
                {
                    if (col0 == row) diag[rt][v] += x0;
                    const float mo = m_run[rt][v];
                    const float e  = __expf(-fabsf(x0 - mo));
                    if (x0 > mo) { s_run[rt][v] = s_run[rt][v] * e + 1.0f; m_run[rt][v] = x0; }
                    else         { s_run[rt][v] += e; }
                }
                {
                    if (col1 == row) diag[rt][v] += x1;
                    const float mo = m_run[rt][v];
                    const float e  = __expf(-fabsf(x1 - mo));
                    if (x1 > mo) { s_run[rt][v] = s_run[rt][v] * e + 1.0f; m_run[rt][v] = x1; }
                    else         { s_run[rt][v] += e; }
                }
            }
        }
    }

    // Combine (m,s,diag) across the 16 lanes that share the same rows.
#pragma unroll
    for (int rt = 0; rt < 2; ++rt) {
#pragma unroll
        for (int v = 0; v < 8; ++v) {
            float m = m_run[rt][v], s = s_run[rt][v], d = diag[rt][v];
#pragma unroll
            for (int mask = 8; mask >= 1; mask >>= 1) {   // inside 16-lane half
                const float mo = __shfl_xor(m, mask, 32);
                const float so = __shfl_xor(s, mask, 32);
                const float dd = __shfl_xor(d, mask, 32);
                const float mn = fmaxf(m, mo);
                s = s * __expf(m - mn) + so * __expf(mo - mn);
                m = mn;
                d += dd;
            }
            m_run[rt][v] = m; s_run[rt][v] = s; diag[rt][v] = d;
        }
    }

    // Lanes 0 and 16 publish the 32 per-row triples for this wave.
    if (l16 == 0) {
#pragma unroll
        for (int rt = 0; rt < 2; ++rt)
#pragma unroll
            for (int v = 0; v < 8; ++v) {
                const int row = rt * 16 + v + 8 * half;
                red_m[wave][row] = m_run[rt][v];
                red_s[wave][row] = s_run[rt][v];
                red_d[wave][row] = diag[rt][v];
            }
    }
    __syncthreads();

    // Combine across waves; one thread per row emits lse - diag.
    if (tid < MROWS) {
        float m = red_m[0][tid], s = red_s[0][tid], d = red_d[0][tid];
#pragma unroll
        for (int w = 1; w < GWAVES; ++w) {
            const float mo = red_m[w][tid], so = red_s[w][tid];
            const float mn = fmaxf(m, mo);
            s = s * __expf(m - mn) + so * __expf(mo - mn);
            m = mn;
            d += red_d[w][tid];
        }
        rowloss[m_base + tid] = (m + __logf(s)) - d;
    }
}

// ---------------------------------------------------------------------------
// Kernel 3: mean over N row losses -> scalar loss.
// ---------------------------------------------------------------------------
__global__ __launch_bounds__(256) void reduce_mean_kernel(
    const float* __restrict__ rowloss, float* __restrict__ out, int N)
{
    __shared__ float sm[8];
    float s = 0.0f;
    for (int i = threadIdx.x; i < N; i += 256) s += rowloss[i];
#pragma unroll
    for (int mask = 16; mask >= 1; mask >>= 1) s += __shfl_xor(s, mask, 32);
    if ((threadIdx.x & 31) == 0) sm[threadIdx.x >> 5] = s;
    __syncthreads();
    if (threadIdx.x == 0) {
        float t = 0.0f;
#pragma unroll
        for (int w = 0; w < 8; ++w) t += sm[w];
        out[0] = t / (float)N;
    }
}

// ---------------------------------------------------------------------------
extern "C" void kernel_launch(void* const* d_in, const int* in_sizes, int n_in,
                              void* d_out, int out_size, void* d_ws, size_t ws_size,
                              hipStream_t stream)
{
    const float* q    = (const float*)d_in[0];   // [N, 768] f32
    const float* p    = (const float*)d_in[1];   // [N, 768] f32
    const float* temp = (const float*)d_in[2];   // scalar f32

    const int N = in_sizes[0] / DIMK;            // 8192

    // Workspace layout: qn f16 | pn f16 | rowloss f32   (~25.2 MB total)
    _Float16* qn      = (_Float16*)d_ws;
    _Float16* pn      = qn + (size_t)N * DIMK;
    float*    rowloss = (float*)(pn + (size_t)N * DIMK);
    float*    out     = (float*)d_out;

    normalize_f16_kernel<<<(2 * N) / 8, 256, 0, stream>>>(q, p, qn, pn, N);
    gemm_lse_kernel<<<N / MROWS, 512, 0, stream>>>(qn, pn, temp, rowloss, N);
    reduce_mean_kernel<<<1, 256, 0, stream>>>(rowloss, out, N);
}
